// SpectralBranch_1125281432181
// MI455X (gfx1250) — matrix operand order repaired and checked by reading, not compiled, not run on hardware
//
#include <hip/hip_runtime.h>
#include <hip/hip_bf16.h>

// ---------------------------------------------------------------------------
// CDNA5 (gfx1250) implementation of the SpectralBranch channel-attention block.
// Heavy GEMMs: v_wmma_f32_16x16x32_bf16 (wave32 WMMA), A-operand staged into
// LDS via the Tensor Data Mover (tensor_load_to_lds + s_wait_tensorcnt).
// ---------------------------------------------------------------------------

typedef __attribute__((ext_vector_type(16))) __bf16       v16bf;
typedef __attribute__((ext_vector_type(8)))  float        v8f;
typedef __attribute__((ext_vector_type(8)))  unsigned int v8u;
typedef __attribute__((ext_vector_type(4)))  unsigned int u32x4;
typedef __attribute__((ext_vector_type(4)))  int          i32x4;
typedef __attribute__((ext_vector_type(8)))  int          i32x8;

#define CCH   384          // channels
#define HW    65536        // 256*256
#define HEADS 8
#define HCH   48           // channels per head

#if __has_builtin(__builtin_amdgcn_tensor_load_to_lds)
  #define USE_TDM 1
  #if __has_include(<hip/amd_detail/amd_gfx1250_TDM.h>)
    #define TDM_ARGS6 1   // therock-10.0 headers -> 6-arg builtin
  #endif
#endif

__device__ __forceinline__ unsigned short f2bf(float f) {
  unsigned int u = __float_as_uint(f);
  u += 0x7FFFu + ((u >> 16) & 1u);      // round-to-nearest-even
  return (unsigned short)(u >> 16);
}
__device__ __forceinline__ float bf2f(unsigned short h) {
  return __uint_as_float(((unsigned int)h) << 16);
}

// ---- WMMA fragment loaders (ISA 7.12.2 layouts, wave32) --------------------
__device__ __forceinline__ v16bf frag_from(const unsigned short* p0,
                                           const unsigned short* p1) {
  uint4 lo = *reinterpret_cast<const uint4*>(p0);
  uint4 hi = *reinterpret_cast<const uint4*>(p1);
  v8u u;
  u[0] = lo.x; u[1] = lo.y; u[2] = lo.z; u[3] = lo.w;
  u[4] = hi.x; u[5] = hi.y; u[6] = hi.z; u[7] = hi.w;
  return __builtin_bit_cast(v16bf, u);
}

// A: row-major [M x K] bf16.  16x32 tile at (m0, k0).
__device__ __forceinline__ v16bf load_frag_A(const unsigned short* A, int ld,
                                             int m0, int k0, int lane) {
  int row = m0 + (lane & 15);
  int kb  = k0 + ((lane & 16) ? 8 : 0);
  const unsigned short* p = A + (size_t)row * (size_t)ld + kb;
  return frag_from(p, p + 16);
}

// B supplied as B^T row-major [N x K] bf16 (so K contiguous). 32x16 tile.
__device__ __forceinline__ v16bf load_frag_B(const unsigned short* Bt, int ld,
                                             int n0, int k0, int lane) {
  int col = n0 + (lane & 15);
  int kb  = k0 + ((lane & 16) ? 8 : 0);
  const unsigned short* p = Bt + (size_t)col * (size_t)ld + kb;
  return frag_from(p, p + 16);
}

// ---------------------------------------------------------------------------
// Kernel 1: per-pixel LayerNorm over channels, output bf16 pixel-major
// xn_t[p*384 + c]  (this is B^T for the QKV GEMM).  Packed b128 stores.
// ---------------------------------------------------------------------------
__global__ void ln_kernel(const float* __restrict__ x,
                          const float* __restrict__ lw,
                          const float* __restrict__ lb,
                          unsigned short* __restrict__ xn_t) {
  int p = blockIdx.x * blockDim.x + threadIdx.x;          // pixel 0..65535
  float s1 = 0.f, s2 = 0.f;
  for (int c = 0; c < CCH; ++c) {
    float v = x[(size_t)c * HW + p];
    s1 += v; s2 += v * v;
  }
  float mu  = s1 * (1.f / CCH);
  float var = fmaxf(s2 * (1.f / CCH) - mu * mu, 0.f);
  float inv = rsqrtf(var + 1e-5f);
  unsigned short* o = xn_t + (size_t)p * CCH;             // 768B pitch: 16B aligned
  for (int c0 = 0; c0 < CCH; c0 += 8) {
    union { unsigned short s[8]; uint4 q; } pk;
#pragma unroll
    for (int j = 0; j < 8; ++j) {
      int c = c0 + j;
      float v = (x[(size_t)c * HW + p] - mu) * inv * lw[c] + lb[c];
      pk.s[j] = f2bf(v);
    }
    *reinterpret_cast<uint4*>(o + c0) = pk.q;
  }
}

// ---------------------------------------------------------------------------
// Kernel 2: f32 -> bf16 elementwise convert (for W_qkv)
// ---------------------------------------------------------------------------
__global__ void cvt_bf16_kernel(const float* __restrict__ in,
                                unsigned short* __restrict__ out, int n) {
  int i = blockIdx.x * blockDim.x + threadIdx.x;
  if (i < n) out[i] = f2bf(in[i]);
}

// ---------------------------------------------------------------------------
// Kernel 3: generic bf16 WMMA GEMM  C[MxN] = A[MxK] * B[KxN]
// A row-major bf16 (K == 384 at both call sites); B given as B^T [N x K] bf16.
// Block = 256 thr = 8 waves; block tile 32(M) x 256(N); wave tile 16 x 64.
// The block's 32xK A-slab (24 KB) is staged into LDS once via TDM, then each
// wave feeds A fragments from LDS while streaming B from global.
// ---------------------------------------------------------------------------
template <bool OUT_BF16>
__global__ void gemm_kernel(const unsigned short* __restrict__ A,
                            const unsigned short* __restrict__ Bt,
                            void* __restrict__ Cout,
                            int M, int N, int K) {
  __shared__ unsigned short sA[32 * 384];                 // 24 KB A slab
  int lane = threadIdx.x & 31;
  int wave = threadIdx.x >> 5;

#if defined(USE_TDM)
  if (wave == 0) {
    // ---- Tensor DMA descriptor (ISA 8.3/8.4): 2D tile [K x 32], 2B elems --
    unsigned long long gaddr =
        (unsigned long long)(uintptr_t)(A + (size_t)blockIdx.y * 32u * (size_t)K);
    unsigned int laddr = (unsigned int)(uintptr_t)&sA[0];
    u32x4 g0;
    g0[0] = 1u;                                           // count=1, user mode
    g0[1] = laddr;                                        // lds_addr
    g0[2] = (unsigned int)(gaddr & 0xFFFFFFFFu);          // global_addr[31:0]
    g0[3] = (unsigned int)((gaddr >> 32) & 0x01FFFFFFu)   // global_addr[56:32]
            | (2u << 30);                                 // type = 2 ("image")
    i32x8 g1;
    g1[0] = (1 << 16);                                    // data_size=1 (2B)
    g1[1] = (int)((K & 0xFFFF) << 16);                    // tensor_dim0[15:0]
    g1[2] = (int)(((K >> 16) & 0xFFFF) | (32u << 16));    // dim0 hi | tensor_dim1=32
    g1[3] = (int)((K & 0xFFFF) << 16);                    // tile_dim0 = K
    g1[4] = 32;                                           // tile_dim1 = 32
    g1[5] = K;                                            // tensor_dim0_stride
    g1[6] = 0;
    g1[7] = 0;
    i32x4 z4 = (i32x4)0;
#if defined(TDM_ARGS6)
    __builtin_amdgcn_tensor_load_to_lds(g0, g1, z4, z4, (i32x8)0, 0);
#else
    __builtin_amdgcn_tensor_load_to_lds(g0, g1, z4, z4, 0);
#endif
    __builtin_amdgcn_s_wait_tensorcnt(0);
  }
  __syncthreads();
#else
  {
    const uint4* src =
        reinterpret_cast<const uint4*>(A + (size_t)blockIdx.y * 32u * (size_t)K);
    uint4* dst = reinterpret_cast<uint4*>(sA);
    for (int i = threadIdx.x; i < 32 * K / 8; i += blockDim.x) dst[i] = src[i];
    __syncthreads();
  }
#endif

  int m_off = (wave & 1) * 16;                            // row in LDS slab
  int m0 = blockIdx.y * 32 + m_off;
  int n0 = blockIdx.x * 256 + (wave >> 1) * 64;

  v8f acc[4];
  acc[0] = {}; acc[1] = {}; acc[2] = {}; acc[3] = {};

  for (int k0 = 0; k0 < K; k0 += 32) {
    v16bf a = load_frag_A(sA, K, m_off, k0, lane);        // from LDS
#pragma unroll
    for (int j = 0; j < 4; ++j) {
      if (k0 + 32 < K) {  // prefetch next B k-slab (global_prefetch_b8)
        __builtin_prefetch(
            Bt + (size_t)(n0 + j * 16 + (lane & 15)) * (size_t)K + k0 + 32, 0, 0);
      }
      v16bf b = load_frag_B(Bt, K, n0 + j * 16, k0, lane);
      acc[j] = __builtin_amdgcn_wmma_f32_16x16x32_bf16(
          false, a, false, b, (short)0, acc[j], false, false);
    }
  }

#pragma unroll
  for (int j = 0; j < 4; ++j) {
#pragma unroll
    for (int r = 0; r < 8; ++r) {
      int row = m0 + r + ((lane & 16) ? 8 : 0);
      int col = n0 + j * 16 + (lane & 15);
      size_t idx = (size_t)row * (size_t)N + col;
      if (OUT_BF16) ((unsigned short*)Cout)[idx] = f2bf(acc[j][r]);
      else          ((float*)Cout)[idx] = acc[j][r];
    }
  }
}

// ---------------------------------------------------------------------------
// Kernel 4: depthwise 3x3 (SAME, zero pad) + spatial gating; scatter q/k/v.
// q,k channel-major bf16 [384 x HW]; v pixel-major bf16 [HW x 384].
// ---------------------------------------------------------------------------
__global__ void dwconv_kernel(const unsigned short* __restrict__ qkv,
                              const float* __restrict__ wdw,
                              const float* __restrict__ sp,
                              unsigned short* __restrict__ q_cm,
                              unsigned short* __restrict__ k_cm,
                              unsigned short* __restrict__ v_pm) {
  int idx = blockIdx.x * blockDim.x + threadIdx.x;        // < 1152*65536
  int c = idx >> 16;
  int n = idx & 65535;
  int py = n >> 8, px = n & 255;
  const unsigned short* base = qkv + (size_t)c * HW;
  const float* w = wdw + c * 9;
  float acc = 0.f;
#pragma unroll
  for (int dy = -1; dy <= 1; ++dy) {
    int yy = py + dy;
    if ((unsigned)yy >= 256u) continue;
#pragma unroll
    for (int dx = -1; dx <= 1; ++dx) {
      int xx = px + dx;
      if ((unsigned)xx >= 256u) continue;
      acc += bf2f(base[yy * 256 + xx]) * w[(dy + 1) * 3 + (dx + 1)];
    }
  }
  acc *= sp[n];
  unsigned short o = f2bf(acc);
  if (c < CCH)            q_cm[(size_t)c * HW + n] = o;
  else if (c < 2 * CCH)   k_cm[(size_t)(c - CCH) * HW + n] = o;
  else                    v_pm[(size_t)n * CCH + (c - 2 * CCH)] = o;
}

// ---------------------------------------------------------------------------
// Kernel 5: per-channel L2 norm of q and k -> 1/max(||.||, 1e-12)
// ---------------------------------------------------------------------------
__global__ void norm_kernel(const unsigned short* __restrict__ q_cm,
                            const unsigned short* __restrict__ k_cm,
                            float* __restrict__ inv_qk) {
  int row = blockIdx.x;                                   // 0..767
  const unsigned short* p = (row < CCH)
      ? q_cm + (size_t)row * HW
      : k_cm + (size_t)(row - CCH) * HW;
  float s = 0.f;
  for (int i = threadIdx.x; i < HW; i += 256) {
    float v = bf2f(p[i]);
    s += v * v;
  }
  __shared__ float sm[256];
  sm[threadIdx.x] = s;
  __syncthreads();
  for (int off = 128; off > 0; off >>= 1) {
    if (threadIdx.x < off) sm[threadIdx.x] += sm[threadIdx.x + off];
    __syncthreads();
  }
  if (threadIdx.x == 0)
    inv_qk[row] = 1.f / fmaxf(sqrtf(sm[0]), 1e-12f);
}

// ---------------------------------------------------------------------------
// Kernel 6: gram G[h][48][48] = q_h * k_h^T via WMMA, split-K over 8 waves.
// grid = (9 tiles, 8 heads), block = 256 threads.
// ---------------------------------------------------------------------------
__global__ void gram_kernel(const unsigned short* __restrict__ q_cm,
                            const unsigned short* __restrict__ k_cm,
                            float* __restrict__ G) {
  int lane = threadIdx.x & 31;
  int wave = threadIdx.x >> 5;
  int head = blockIdx.y;
  int tile = blockIdx.x;                  // 0..8 -> 3x3 grid of 16x16 tiles
  int tm = (tile / 3) * 16;
  int tn = (tile % 3) * 16;
  const unsigned short* A = q_cm + (size_t)head * HCH * HW;
  const unsigned short* B = k_cm + (size_t)head * HCH * HW;

  v8f acc = {};
  int kstart = wave * (HW / 8);
  for (int k0 = kstart; k0 < kstart + HW / 8; k0 += 32) {
    v16bf a = load_frag_A(A, HW, tm, k0, lane);
    v16bf b = load_frag_B(B, HW, tn, k0, lane);
    acc = __builtin_amdgcn_wmma_f32_16x16x32_bf16(
        false, a, false, b, (short)0, acc, false, false);
  }

  __shared__ float red[8 * 256];
#pragma unroll
  for (int r = 0; r < 8; ++r) red[wave * 256 + lane * 8 + r] = acc[r];
  __syncthreads();

  int t = threadIdx.x;                    // 0..255 : flat = lane*8 + r
  float s = 0.f;
#pragma unroll
  for (int w = 0; w < 8; ++w) s += red[w * 256 + t];
  int lane_o = t >> 3;
  int r = t & 7;
  int Mrow = tm + r + ((lane_o & 16) ? 8 : 0);
  int Ncol = tn + (lane_o & 15);
  G[head * (HCH * HCH) + Mrow * HCH + Ncol] = s;
}

// ---------------------------------------------------------------------------
// Kernel 7: scale by 1/||q||, 1/||k||, temperature; softmax over 48 cols.
// ---------------------------------------------------------------------------
__global__ void softmax_kernel(const float* __restrict__ G,
                               const float* __restrict__ inv_qk,
                               const float* __restrict__ temp,
                               float* __restrict__ attn) {
  int idx = blockIdx.x * blockDim.x + threadIdx.x;
  if (idx >= HEADS * HCH) return;
  int h = idx / HCH;
  int cl = idx % HCH;
  const float* g = G + h * (HCH * HCH) + cl * HCH;
  const float* ik = inv_qk + CCH + h * HCH;
  float iq = inv_qk[idx];
  float t = temp[h];
  float l[HCH];
  float mx = -1e30f;
  for (int d = 0; d < HCH; ++d) {
    l[d] = g[d] * iq * ik[d] * t;
    mx = fmaxf(mx, l[d]);
  }
  float s = 0.f;
  for (int d = 0; d < HCH; ++d) { l[d] = __expf(l[d] - mx); s += l[d]; }
  float r = 1.f / s;
  float* o = attn + h * (HCH * HCH) + cl * HCH;
  for (int d = 0; d < HCH; ++d) o[d] = l[d] * r;
}

// ---------------------------------------------------------------------------
// Kernel 8: fold attention into the projection:
//   Mmat[o][dg] = sum_c W_proj[o][h(dg)*48+c] * attn[h][c][dg%48]   (bf16 out)
// ---------------------------------------------------------------------------
__global__ void mmat_kernel(const float* __restrict__ wproj,
                            const float* __restrict__ attn,
                            unsigned short* __restrict__ Mm) {
  int idx = blockIdx.x * blockDim.x + threadIdx.x;        // < 384*384
  int o = idx / CCH;
  int dg = idx % CCH;
  int h = dg / HCH;
  int dl = dg % HCH;
  const float* wp = wproj + o * CCH + h * HCH;
  const float* a  = attn + h * (HCH * HCH) + dl;          // stride 48 over c
  float s = 0.f;
#pragma unroll
  for (int c = 0; c < HCH; ++c) s += wp[c] * a[c * HCH];
  Mm[idx] = f2bf(s);
}

// ---------------------------------------------------------------------------
extern "C" void kernel_launch(void* const* d_in, const int* in_sizes, int n_in,
                              void* d_out, int out_size, void* d_ws, size_t ws_size,
                              hipStream_t stream) {
  const float* x    = (const float*)d_in[0];
  const float* sp   = (const float*)d_in[1];
  const float* lw   = (const float*)d_in[2];
  const float* lb   = (const float*)d_in[3];
  const float* wqkv = (const float*)d_in[4];
  const float* wdw  = (const float*)d_in[5];
  const float* wprj = (const float*)d_in[6];
  const float* temp = (const float*)d_in[7];
  float* out = (float*)d_out;

  // workspace carve-up (256B aligned)
  char* ws = (char*)d_ws;
  size_t off = 0;
  auto carve = [&](size_t bytes) -> void* {
    void* p = ws + off;
    off = (off + bytes + 255) & ~(size_t)255;
    return p;
  };
  unsigned short* xn_t   = (unsigned short*)carve((size_t)HW * CCH * 2);       // LN out, pixel-major
  unsigned short* wqkv_b = (unsigned short*)carve((size_t)3 * CCH * CCH * 2);  // bf16 W_qkv
  unsigned short* qkv_b  = (unsigned short*)carve((size_t)3 * CCH * HW * 2);   // GEMM1 out
  unsigned short* q_cm   = (unsigned short*)carve((size_t)CCH * HW * 2);
  unsigned short* k_cm   = (unsigned short*)carve((size_t)CCH * HW * 2);
  unsigned short* v_pm   = (unsigned short*)carve((size_t)HW * CCH * 2);
  float* inv_qk = (float*)carve((size_t)2 * CCH * 4);
  float* G      = (float*)carve((size_t)HEADS * HCH * HCH * 4);
  float* attn   = (float*)carve((size_t)HEADS * HCH * HCH * 4);
  unsigned short* Mmat = (unsigned short*)carve((size_t)CCH * CCH * 4);

  // 1) LayerNorm -> bf16 pixel-major
  ln_kernel<<<HW / 256, 256, 0, stream>>>(x, lw, lb, xn_t);

  // 2) W_qkv -> bf16
  cvt_bf16_kernel<<<(3 * CCH * CCH + 255) / 256, 256, 0, stream>>>(
      wqkv, wqkv_b, 3 * CCH * CCH);

  // 3) qkv = W_qkv[1152x384] * xn[384x65536]  (TDM + WMMA, bf16 out)
  {
    dim3 grid(HW / 256, (3 * CCH) / 32);
    gemm_kernel<true><<<grid, 256, 0, stream>>>(wqkv_b, xn_t, qkv_b,
                                                3 * CCH, HW, CCH);
  }

  // 4) depthwise 3x3 + spatial gate; layout scatter for the next GEMMs
  dwconv_kernel<<<(3 * CCH * HW) / 256, 256, 0, stream>>>(
      qkv_b, wdw, sp, q_cm, k_cm, v_pm);

  // 5) per-channel inverse L2 norms of q, k
  norm_kernel<<<2 * CCH, 256, 0, stream>>>(q_cm, k_cm, inv_qk);

  // 6) raw gram per head via WMMA (split-K across 8 waves)
  {
    dim3 grid(9, HEADS);
    gram_kernel<<<grid, 256, 0, stream>>>(q_cm, k_cm, G);
  }

  // 7) scale + softmax
  softmax_kernel<<<(HEADS * HCH + 255) / 256, 256, 0, stream>>>(
      G, inv_qk, temp, attn);

  // 8) fold attention into projection matrix
  mmat_kernel<<<(CCH * CCH) / 256, 256, 0, stream>>>(wprj, attn, Mmat);

  // 9) out = Mmat[384x384] * v[384x65536]  (TDM + WMMA, f32 out -> d_out)
  {
    dim3 grid(HW / 256, CCH / 32);
    gemm_kernel<false><<<grid, 256, 0, stream>>>(Mmat, v_pm, (void*)out,
                                                 CCH, HW, CCH);
  }
}
